// LinearAttention_24464133718508
// MI455X (gfx1250) — matrix operand order repaired and checked by reading, not compiled, not run on hardware
//
#include <hip/hip_runtime.h>

// ---------------- problem constants (from reference) ----------------
#define HEADS   4
#define DHEAD   32
#define DIMC    512          // DIM
#define HIDDEN  128          // HEADS*DIM_HEAD
#define LSEQ    8192
#define BATCH   8
#define NSLICE  8            // K-slices for the context reduction

typedef __attribute__((ext_vector_type(16))) __bf16 v16bf;
typedef __attribute__((ext_vector_type(8)))  __bf16 v8bf;
typedef __attribute__((ext_vector_type(8)))  float  v8f;

// pack two f32 into one dword of bf16 (round-to-nearest-ish)
static __device__ __forceinline__ unsigned pack2(float a, float b) {
    unsigned ua = (__builtin_bit_cast(unsigned, a) + 0x8000u) >> 16;
    unsigned ub = (__builtin_bit_cast(unsigned, b) + 0x8000u) & 0xFFFF0000u;
    return ua | ub;
}

static __device__ __forceinline__ v16bf frag_cat(v8bf lo, v8bf hi) {
    return __builtin_shufflevector(lo, hi, 0,1,2,3,4,5,6,7,8,9,10,11,12,13,14,15);
}

// =====================================================================
// Generic fp32-in / fp32-out GEMM, bf16 WMMA compute, f32 accumulate.
//   C[b] (M x 8192) = A[b] (M x K, row-major, lda) @ Bm[b] (K x 8192) [+ bias]
// Block: 256 threads = 8 waves in a 4(M) x 2(N) grid; block tile 128x128;
// each wave computes 32(M) x 64(N) = 2 A-frags x 4 B-frags -> 8 WMMA/k-step.
// Double-buffered bf16 LDS tiles in fragment-native layout (stride 40):
//   As: row-major [row][K]; Bs: transposed [col][K].
// One barrier per k-step; staging pointers are loop-carried (no per-step
// 64-bit address rebuilds).
// grid = (8192/128, M/128, BATCH)
// =====================================================================
__global__ __launch_bounds__(256)
void gemm_bf16_kernel(const float* __restrict__ A, long a_bstride, int lda,
                      const float* __restrict__ Bm, long b_bstride,
                      float* __restrict__ C, long c_bstride,
                      const float* __restrict__ bias, int K)
{
    const int   n_base = blockIdx.x * 128;
    const int   m_base = blockIdx.y * 128;
    const long  bi     = blockIdx.z;

    const float* Ab = A  + bi * a_bstride;
    const float* Bb = Bm + bi * b_bstride;
    float*       Cb = C  + bi * c_bstride;

    __shared__ __bf16 As[2][128 * 40];
    __shared__ __bf16 Bs[2][128 * 40];

    const int tid   = threadIdx.x;
    const int wave  = tid >> 5;
    const int lane  = tid & 31;
    const int lhalf = lane & 15;
    const bool hi   = lane >= 16;
    const int mrow  = (wave & 3) * 32;   // wave's M offset inside tile
    const int ncol  = (wave >> 2) * 64;  // wave's N offset inside tile

    // ---- loop-carried staging pointers ----
    const float* pA[4];  int dA[4];
    #pragma unroll
    for (int it = 0; it < 4; ++it) {
        int lin = tid + it * 256;                  // 1024 float4 units
        int r   = lin >> 3;
        int c4  = (lin & 7) << 2;
        pA[it] = Ab + (long)(m_base + r) * lda + c4;
        dA[it] = r * 40 + c4;
    }
    const float* pB[8];  int dB[8];
    #pragma unroll
    for (int it = 0; it < 8; ++it) {
        int u  = tid + it * 256;                   // 2048 dword units
        int kp = u >> 7;
        int n  = u & 127;
        pB[it] = Bb + (long)(2 * kp) * LSEQ + n_base + n;
        dB[it] = n * 40 + 2 * kp;
    }

    auto stage = [&](int buf) {
        #pragma unroll
        for (int it = 0; it < 4; ++it) {
            float4 v = *(const float4*)pA[it];
            uint2 p; p.x = pack2(v.x, v.y); p.y = pack2(v.z, v.w);
            *(uint2*)(&As[buf][dA[it]]) = p;       // ds_store_b64
            pA[it] += 32;
        }
        #pragma unroll
        for (int it = 0; it < 8; ++it) {
            unsigned d = pack2(pB[it][0], pB[it][LSEQ]);
            *(unsigned*)(&Bs[buf][dB[it]]) = d;    // ds_store_b32
            pB[it] += 32L * LSEQ;
        }
    };

    v8f acc[2][4] = {};
    const int kb    = hi ? 8 : 0;
    const int khalf = hi ? 16 : 0;
    const int nsteps = K / 32;

    stage(0);

    for (int s = 0; s < nsteps; ++s) {
        const int buf = s & 1;
        __syncthreads();                           // staging of buf complete
        if (s + 1 < nsteps) stage(buf ^ 1);        // overlap next tile
        if (s + 2 < nsteps) {                      // prefetch tile s+2
            __builtin_prefetch(pA[0], 0, 1);
            __builtin_prefetch(pB[0], 0, 1);
        }

        v16bf af[2];
        #pragma unroll
        for (int i = 0; i < 2; ++i) {
            const __bf16* ap = &As[buf][(mrow + i * 16 + lhalf) * 40 + kb];
            af[i] = frag_cat(*(const v8bf*)ap, *(const v8bf*)(ap + 16));
        }
        #pragma unroll
        for (int j = 0; j < 4; ++j) {
            const __bf16* bp = &Bs[buf][(ncol + j * 16 + lhalf) * 40 + kb];
            v16bf bf = frag_cat(*(const v8bf*)bp, *(const v8bf*)(bp + 16));
            #pragma unroll
            for (int i = 0; i < 2; ++i)
                acc[i][j] = __builtin_amdgcn_wmma_f32_16x16x32_bf16(
                    false, af[i], false, bf, (short)0, acc[i][j], false, false);
        }
    }

    // ---- store C per 16x16 f32 C/D layout ----
    #pragma unroll
    for (int i = 0; i < 2; ++i) {
        const int rbase = m_base + mrow + i * 16 + (hi ? 8 : 0);
        #pragma unroll
        for (int j = 0; j < 4; ++j) {
            const int col = n_base + ncol + j * 16 + lhalf;
            #pragma unroll
            for (int r = 0; r < 8; ++r) {
                const int row = rbase + r;
                float v = acc[i][j][r];
                if (bias) v += bias[row];
                Cb[(long)row * LSEQ + col] = v;
            }
        }
    }
}

// =====================================================================
// Softmax over L for each k row (in place). qkv: [b][384][L]; k rows 128..255.
// grid = BATCH*HIDDEN blocks, 256 threads, float4 streaming.
// =====================================================================
__global__ __launch_bounds__(256)
void softmax_rows_kernel(float* __restrict__ qkv)
{
    const long row = blockIdx.x;
    float* p = qkv + (row >> 7) * (3L * HIDDEN * LSEQ)
                   + (long)(HIDDEN + (row & 127)) * LSEQ;
    float4* p4 = (float4*)p;

    __shared__ float red[256];
    const int t = threadIdx.x;

    float m = -3.402823466e38f;
    #pragma unroll
    for (int i = t; i < LSEQ / 4; i += 256) {
        float4 v = p4[i];
        m = fmaxf(m, fmaxf(fmaxf(v.x, v.y), fmaxf(v.z, v.w)));
    }
    red[t] = m; __syncthreads();
    for (int s = 128; s > 0; s >>= 1) {
        if (t < s) red[t] = fmaxf(red[t], red[t + s]);
        __syncthreads();
    }
    m = red[0]; __syncthreads();

    float sum = 0.f;
    #pragma unroll
    for (int i = t; i < LSEQ / 4; i += 256) {
        float4 v = p4[i];
        v.x = __expf(v.x - m); v.y = __expf(v.y - m);
        v.z = __expf(v.z - m); v.w = __expf(v.w - m);
        p4[i] = v;
        sum += (v.x + v.y) + (v.z + v.w);
    }
    red[t] = sum; __syncthreads();
    for (int s = 128; s > 0; s >>= 1) {
        if (t < s) red[t] += red[t + s];
        __syncthreads();
    }
    const float inv = 1.f / red[0];
    #pragma unroll
    for (int i = t; i < LSEQ / 4; i += 256) {
        float4 v = p4[i];
        v.x *= inv; v.y *= inv; v.z *= inv; v.w *= inv;
        p4[i] = v;
    }
}

// =====================================================================
// Partial context: part[sl][b][h][d][e] = sum_{n in slice} k[d,n]*v[e,n]
// grid = (BATCH*HEADS, NSLICE); block = 128 (4 waves, one 16x16 tile each).
// Double-buffered bf16 tiles; v[e][n] is already B-fragment-native [col][K].
// =====================================================================
__global__ __launch_bounds__(128)
void ctx_kernel(const float* __restrict__ qkv, float* __restrict__ part)
{
    const int bh = blockIdx.x;
    const int sl = blockIdx.y;
    const int b  = bh >> 2;
    const int h  = bh & 3;
    const float* kbase = qkv + (long)b * 3 * HIDDEN * LSEQ + (long)(HIDDEN     + h * DHEAD) * LSEQ;
    const float* vbase = qkv + (long)b * 3 * HIDDEN * LSEQ + (long)(2 * HIDDEN + h * DHEAD) * LSEQ;

    __shared__ __bf16 Ks[2][32 * 40];
    __shared__ __bf16 Vs[2][32 * 40];

    const int tid   = threadIdx.x;
    const int wave  = tid >> 5;
    const int lane  = tid & 31;
    const int lhalf = lane & 15;
    const bool hi   = lane >= 16;
    const int dt = (wave >> 1) * 16;
    const int et = (wave & 1) * 16;

    const int n_begin = sl * (LSEQ / NSLICE);
    const int nsteps  = (LSEQ / NSLICE) / 32;

    // loop-carried staging pointers: 2 float4 units per tile per thread
    const float* pK[2]; const float* pV[2]; int dKV[2];
    #pragma unroll
    for (int it = 0; it < 2; ++it) {
        int lin = tid + it * 128;                  // 256 float4 units
        int r   = lin >> 3;
        int c4  = (lin & 7) << 2;
        pK[it]  = kbase + (long)r * LSEQ + n_begin + c4;
        pV[it]  = vbase + (long)r * LSEQ + n_begin + c4;
        dKV[it] = r * 40 + c4;
    }

    auto stage = [&](int buf) {
        #pragma unroll
        for (int it = 0; it < 2; ++it) {
            float4 a = *(const float4*)pK[it];
            float4 c = *(const float4*)pV[it];
            uint2 pa; pa.x = pack2(a.x, a.y); pa.y = pack2(a.z, a.w);
            uint2 pc; pc.x = pack2(c.x, c.y); pc.y = pack2(c.z, c.w);
            *(uint2*)(&Ks[buf][dKV[it]]) = pa;
            *(uint2*)(&Vs[buf][dKV[it]]) = pc;
            pK[it] += 32; pV[it] += 32;
        }
    };

    v8f acc = {};
    const int kb    = hi ? 8 : 0;
    const int khalf = hi ? 16 : 0;

    stage(0);
    for (int s = 0; s < nsteps; ++s) {
        const int buf = s & 1;
        __syncthreads();
        if (s + 1 < nsteps) stage(buf ^ 1);

        const __bf16* ap = &Ks[buf][(dt + lhalf) * 40 + kb];
        v16bf afrag = frag_cat(*(const v8bf*)ap, *(const v8bf*)(ap + 16));
        const __bf16* bp = &Vs[buf][(et + lhalf) * 40 + kb];
        v16bf bfrag = frag_cat(*(const v8bf*)bp, *(const v8bf*)(bp + 16));

        acc = __builtin_amdgcn_wmma_f32_16x16x32_bf16(
            false, afrag, false, bfrag, (short)0, acc, false, false);
    }

    float* dst = part + ((long)sl * (BATCH * HEADS) + bh) * DHEAD * DHEAD;
    const int dbase = dt + (hi ? 8 : 0);
    #pragma unroll
    for (int r = 0; r < 8; ++r)
        dst[(dbase + r) * DHEAD + et + lhalf] = acc[r];
}

// =====================================================================
// ctx[bh][d][e] = sum_sl part[sl][bh][d][e]   (deterministic reduce)
// =====================================================================
__global__ __launch_bounds__(256)
void ctx_reduce_kernel(const float* __restrict__ part, float* __restrict__ ctx)
{
    const int idx = blockIdx.x * 256 + threadIdx.x;
    const int tot = BATCH * HEADS * DHEAD * DHEAD;
    if (idx >= tot) return;
    float s = 0.f;
    #pragma unroll
    for (int sl = 0; sl < NSLICE; ++sl) s += part[(long)sl * tot + idx];
    ctx[idx] = s;
}

// =====================================================================
// W2[b][o][h*32+d] = sum_e w_out[o][h*32+e] * ctx[b][h][d][e]
// =====================================================================
__global__ __launch_bounds__(256)
void w2_kernel(const float* __restrict__ w_out, const float* __restrict__ ctx,
               float* __restrict__ W2)
{
    const long idx = (long)blockIdx.x * 256 + threadIdx.x;
    if (idx >= (long)BATCH * DIMC * HIDDEN) return;
    const int c = (int)(idx % HIDDEN);
    const int o = (int)((idx / HIDDEN) % DIMC);
    const int b = (int)(idx / ((long)HIDDEN * DIMC));
    const int h = c >> 5;
    const int d = c & 31;
    const float* wrow = w_out + (long)o * HIDDEN + h * DHEAD;
    const float* crow = ctx + (((long)b * HEADS + h) * DHEAD + d) * DHEAD;
    float s = 0.f;
    #pragma unroll
    for (int e = 0; e < DHEAD; ++e) s += wrow[e] * crow[e];
    W2[idx] = s;
}

// =====================================================================
extern "C" void kernel_launch(void* const* d_in, const int* in_sizes, int n_in,
                              void* d_out, int out_size, void* d_ws, size_t ws_size,
                              hipStream_t stream)
{
    const float* x     = (const float*)d_in[0];   // (8, 512, 8192)
    const float* w_qkv = (const float*)d_in[1];   // (384, 512)
    const float* w_out = (const float*)d_in[2];   // (512, 128)
    const float* b_out = (const float*)d_in[3];   // (512,)
    float* out = (float*)d_out;                   // (8, 512, 8192)

    float* qkv  = (float*)d_ws;                               // 8*384*8192
    float* part = qkv  + (size_t)BATCH * 3 * HIDDEN * LSEQ;   // NSLICE*32*32*32
    float* ctx  = part + (size_t)NSLICE * BATCH * HEADS * DHEAD * DHEAD;
    float* W2   = ctx  + (size_t)BATCH * HEADS * DHEAD * DHEAD;

    // 1) qkv[b] = w_qkv @ x[b]     (M=384, K=512)
    {
        dim3 g(LSEQ / 128, 384 / 128, BATCH);
        gemm_bf16_kernel<<<g, 256, 0, stream>>>(
            w_qkv, 0, DIMC,
            x, (long)DIMC * LSEQ,
            qkv, (long)3 * HIDDEN * LSEQ,
            nullptr, DIMC);
    }

    // 2) softmax over L on k rows (in place)
    softmax_rows_kernel<<<BATCH * HIDDEN, 256, 0, stream>>>(qkv);

    // 3) partial contexts, then reduce
    {
        dim3 g(BATCH * HEADS, NSLICE);
        ctx_kernel<<<g, 128, 0, stream>>>(qkv, part);
        int tot = BATCH * HEADS * DHEAD * DHEAD;
        ctx_reduce_kernel<<<(tot + 255) / 256, 256, 0, stream>>>(part, ctx);
    }

    // 4a) W2 = w_out @ blockdiag(ctx^T)
    {
        long tot = (long)BATCH * DIMC * HIDDEN;
        w2_kernel<<<(unsigned)((tot + 255) / 256), 256, 0, stream>>>(w_out, ctx, W2);
    }

    // 4b) out[b] = W2[b] @ q[b] + b_out   (M=512, K=128; q = qkv rows 0..127)
    {
        dim3 g(LSEQ / 128, DIMC / 128, BATCH);
        gemm_bf16_kernel<<<g, 256, 0, stream>>>(
            W2, (long)DIMC * HIDDEN, HIDDEN,
            qkv, (long)3 * HIDDEN * LSEQ,
            out, (long)DIMC * LSEQ,
            b_out, HIDDEN);
    }
}
